// SimpleMambaBlock_8572754723183
// MI455X (gfx1250) — matrix operand order repaired and checked by reading, not compiled, not run on hardware
//
#include <hip/hip_runtime.h>
#include <hip/hip_bf16.h>
#include <math.h>

// ---------------------------------------------------------------------------
// SimpleMambaBlock for MI455X (gfx1250, wave32, WMMA)
//   GEMMs: bf16 in, fp32 accumulate, v_wmma_f32_16x16x32_bf16.
//          All B operands are weights, pre-transposed at convert time so both
//          A and B stage as contiguous 16B rows. Staging uses the CDNA5 async
//          global->LDS path (ASYNCcnt) when available, double-buffered LDS.
//   Scan : 16 lanes per channel (one SSM state each) + half-wave shfl_xor
// ---------------------------------------------------------------------------

typedef __attribute__((ext_vector_type(16))) __bf16 v16bf;
typedef __attribute__((ext_vector_type(8)))  float  v8f;
typedef __attribute__((ext_vector_type(4)))  int    v4i;

#define AS1 __attribute__((address_space(1)))
#define AS3 __attribute__((address_space(3)))

#if __has_builtin(__builtin_amdgcn_global_load_async_to_lds_b128)
#define HAVE_ASYNC_LDS 1
#else
#define HAVE_ASYNC_LDS 0
#endif

__device__ __forceinline__ void async_wait_zero() {
#if HAVE_ASYNC_LDS
#if __has_builtin(__builtin_amdgcn_s_wait_asynccnt)
  __builtin_amdgcn_s_wait_asynccnt(0);
#else
  asm volatile("s_wait_asynccnt 0x0" ::: "memory");
#endif
#endif
}

#define D_MODEL 1024
#define D_STATE 16
#define D_CONV  4
#define D_INNER 2048
#define DT_RANK 64
#define BATCH   2
#define SEQ     1024
#define NTOK    (BATCH*SEQ)            // 2048 token rows
#define P_DIM   (DT_RANK + 2*D_STATE)  // 96 (logical)
#define P_LD    128                    // padded leading dim for ssm projections

// ---------------- f32 -> bf16 convert, 4-wide (grid-stride) ----------------
__global__ __launch_bounds__(256)
void cvt_f32_bf16x4_kernel(__bf16* __restrict__ dst, const float* __restrict__ src, int n4) {
  union Pack { __bf16 h[4]; uint2 u; };
  int i = blockIdx.x * 256 + threadIdx.x;
  int stride = gridDim.x * 256;
  for (; i < n4; i += stride) {
    float4 f = reinterpret_cast<const float4*>(src)[i];
    Pack p;
    p.h[0] = (__bf16)f.x; p.h[1] = (__bf16)f.y;
    p.h[2] = (__bf16)f.z; p.h[3] = (__bf16)f.w;
    reinterpret_cast<uint2*>(dst)[i] = p.u;
  }
}

// -------- transpose + convert: dst[n][k] = bf16(src[k][n]), zero-pad n >= Nsrc
// src: K x Nsrc f32 (row-major); dst: Ndst x K bf16. grid = (Ndst/32, K/32).
__global__ __launch_bounds__(256)
void transpose_cvt_kernel(__bf16* __restrict__ dst, const float* __restrict__ src,
                          int K, int Nsrc) {
  __shared__ float tile[32][33];
  int n0 = blockIdx.x * 32;
  int k0 = blockIdx.y * 32;
  int tx = threadIdx.x & 31, ty = threadIdx.x >> 5;   // 32 x 8
#pragma unroll
  for (int i = 0; i < 4; ++i) {
    int n = n0 + tx, k = k0 + ty + i * 8;
    tile[ty + i * 8][tx] = (n < Nsrc) ? src[(size_t)k * Nsrc + n] : 0.f;
  }
  __syncthreads();
#pragma unroll
  for (int i = 0; i < 4; ++i) {
    int n = n0 + ty + i * 8, k = k0 + tx;
    dst[(size_t)n * K + k] = (__bf16)tile[tx][ty + i * 8];
  }
}

// ---------------- bf16 GEMM, fp32 accumulate, WMMA 16x16x32 ----------------
// C[M x N] (f32, ldc) = A[M x K] (bf16, lda) * Bt[N x K] (bf16, ldbt; B transposed)
// Exact tiles only: M % 128 == 0, N % 128 == 0, K % 32 == 0.
// Block: 256 thr = 8 waves (4 in M x 2 in N); block tile 128x128; wave tile 32x64.
// Double-buffered LDS; staging via global_load_async_to_lds_b128 when available.
#define BM 128
#define BN 128
#define BK 32

__global__ __launch_bounds__(256)
void gemm_bf16_wmma_kernel(const __bf16* __restrict__ A, int lda,
                           const __bf16* __restrict__ Bt, int ldbt,
                           float* __restrict__ C, int ldc,
                           int M, int N, int K)
{
  // +8 halves pad => row stride 80 B (multiple of 16: b128 LDS ops stay aligned)
  __shared__ __bf16 sA [2][BM][BK + 8];   // [m][k]
  __shared__ __bf16 sBt[2][BN][BK + 8];   // [n][k]

  const int tid  = threadIdx.x;
  const int lane = tid & 31;
  const int wave = tid >> 5;     // 0..7
  const int wm   = wave >> 1;    // 0..3 : M direction (32 rows each)
  const int wn   = wave & 1;     // 0..1 : N direction (64 cols each)
  const int lr   = lane & 15;
  const int lh   = lane >> 4;    // half-wave select

  const int m0 = blockIdx.y * BM;
  const int n0 = blockIdx.x * BN;

  v8f acc[2][4];
  v8f zero = {0.f,0.f,0.f,0.f,0.f,0.f,0.f,0.f};
#pragma unroll
  for (int mt = 0; mt < 2; ++mt)
#pragma unroll
    for (int nt = 0; nt < 4; ++nt) acc[mt][nt] = zero;

  union Frag { uint4 q[2]; v16bf v; };

  // Both tiles are 128 rows x 32 halves; 2 x 16B per thread per tile.
  auto stage = [&](int buf, int k0) {
#pragma unroll
    for (int p = 0; p < 2; ++p) {
      int flat = tid + p * 256;
      int r  = flat >> 2;           // 4 threads per row
      int c8 = (flat & 3) << 3;     // 8 halves each
#if HAVE_ASYNC_LDS
      __builtin_amdgcn_global_load_async_to_lds_b128(
          (AS1 v4i*)(A + (size_t)(m0 + r) * lda + (k0 + c8)),
          (AS3 v4i*)&sA[buf][r][c8], 0, 0);
      __builtin_amdgcn_global_load_async_to_lds_b128(
          (AS1 v4i*)(Bt + (size_t)(n0 + r) * ldbt + (k0 + c8)),
          (AS3 v4i*)&sBt[buf][r][c8], 0, 0);
#else
      *reinterpret_cast<uint4*>(&sA[buf][r][c8]) =
          *reinterpret_cast<const uint4*>(A + (size_t)(m0 + r) * lda + (k0 + c8));
      *reinterpret_cast<uint4*>(&sBt[buf][r][c8]) =
          *reinterpret_cast<const uint4*>(Bt + (size_t)(n0 + r) * ldbt + (k0 + c8));
#endif
    }
  };

  const int nIter = K / BK;
  stage(0, 0);

  for (int it = 0; it < nIter; ++it) {
    const int buf = it & 1;
    async_wait_zero();     // staged data for `buf` resident in LDS
    __syncthreads();
    if (it + 1 < nIter) stage(buf ^ 1, (it + 1) * BK);  // in flight during WMMAs

    // ---- fragments per documented CDNA5 16-bit layouts ----
    // A 16x32: lane (lr,lh): halves 0..7 = K lh*8..lh*8+7 ; halves 8..15 = +16
    Frag af[2];
#pragma unroll
    for (int mt = 0; mt < 2; ++mt) {
      const __bf16* pa = &sA[buf][wm * 32 + mt * 16 + lr][lh * 8];
      af[mt].q[0] = *reinterpret_cast<const uint4*>(pa);
      af[mt].q[1] = *reinterpret_cast<const uint4*>(pa + 16);
    }
    // B 32x16: lane (lr,lh): halves 0..7 = K lh*8..lh*8+7 ; halves 8..15 = +16
    Frag bfr[4];
#pragma unroll
    for (int nt = 0; nt < 4; ++nt) {
      const __bf16* pb = &sBt[buf][wn * 64 + nt * 16 + lr][lh * 8];
      bfr[nt].q[0] = *reinterpret_cast<const uint4*>(pb);
      bfr[nt].q[1] = *reinterpret_cast<const uint4*>(pb + 16);
    }
#pragma unroll
    for (int mt = 0; mt < 2; ++mt)
#pragma unroll
      for (int nt = 0; nt < 4; ++nt)
        acc[mt][nt] = __builtin_amdgcn_wmma_f32_16x16x32_bf16(
            false, af[mt].v, false, bfr[nt].v,
            (short)0, acc[mt][nt], false, false);
  }

  // ---- store: C/D layout: lane col = lr, VGPR j -> row lh*8 + j ----
#pragma unroll
  for (int mt = 0; mt < 2; ++mt)
#pragma unroll
    for (int nt = 0; nt < 4; ++nt) {
      int gc = n0 + wn * 64 + nt * 16 + lr;
#pragma unroll
      for (int j = 0; j < 8; ++j) {
        int gr = m0 + wm * 32 + mt * 16 + lh * 8 + j;
        C[(size_t)gr * ldc + gc] = acc[mt][nt][j];
      }
    }
}

// ---------------- depthwise causal conv (4-tap) + SiLU ----------------
__global__ __launch_bounds__(256)
void conv_silu_kernel(const float* __restrict__ xz,
                      const float* __restrict__ cw, const float* __restrict__ cb,
                      float* __restrict__ xa, __bf16* __restrict__ xa_bf) {
  int idx = blockIdx.x * 256 + threadIdx.x;
  if (idx >= NTOK * D_INNER) return;
  int c   = idx & (D_INNER - 1);
  int row = idx >> 11;
  int s   = row & (SEQ - 1);
  int b   = row >> 10;
  float acc = cb[c];
#pragma unroll
  for (int j = 0; j < D_CONV; ++j) {
    int sj = s + j - (D_CONV - 1);
    if (sj >= 0)
      acc += xz[((size_t)(b * SEQ + sj)) * (2 * D_INNER) + c] * cw[c * D_CONV + j];
  }
  float v = acc / (1.f + __expf(-acc));  // silu
  xa[idx]    = v;
  xa_bf[idx] = (__bf16)v;
}

// ---------------- dt = softplus(dt_lin + b_dt), in place ----------------
__global__ __launch_bounds__(256)
void softplus_bias_kernel(float* __restrict__ dt, const float* __restrict__ b_dt) {
  int idx = blockIdx.x * 256 + threadIdx.x;
  if (idx >= NTOK * D_INNER) return;
  int c = idx & (D_INNER - 1);
  float t = dt[idx] + b_dt[c];
  dt[idx] = (t > 20.f) ? t : log1pf(__expf(t));
}

// ---------------- selective scan ----------------
__global__ __launch_bounds__(256)
void scan_kernel(const float* __restrict__ dt, const float* __restrict__ xa,
                 const float* __restrict__ ssm_p, const float* __restrict__ A_log,
                 float* __restrict__ y, float* __restrict__ h_last) {
  int t    = blockIdx.x * 256 + threadIdx.x;
  int j    = t & (D_STATE - 1);
  int chan = t >> 4;                       // 0 .. BATCH*D_INNER-1
  if (chan >= BATCH * D_INNER) return;
  int b = chan >> 11;
  int n = chan & (D_INNER - 1);

  float a = -__expf(A_log[n * D_STATE + j]);
  const float* dt_bn = dt + (size_t)b * SEQ * D_INNER + n;
  const float* xa_bn = xa + (size_t)b * SEQ * D_INNER + n;
  const float* p_b   = ssm_p + (size_t)b * SEQ * P_LD;     // padded ld = 128
  float*       y_bn  = y + (size_t)b * SEQ * D_INNER + n;

  float h = 0.f;
  for (int s = 0; s < SEQ; ++s) {
    float dts = dt_bn[(size_t)s * D_INNER];
    float xs  = xa_bn[(size_t)s * D_INNER];
    float Bj  = p_b[s * P_LD + DT_RANK + j];
    float Cj  = p_b[s * P_LD + DT_RANK + D_STATE + j];
    h = __expf(dts * a) * h + (dts * xs) * Bj;
    float yy = h * Cj;
    yy += __shfl_xor(yy, 1, 16);
    yy += __shfl_xor(yy, 2, 16);
    yy += __shfl_xor(yy, 4, 16);
    yy += __shfl_xor(yy, 8, 16);
    if (j == 0) y_bn[(size_t)s * D_INNER] = yy;
  }
  h_last[(size_t)chan * D_STATE + j] = h;   // (B, D_INNER, D_STATE) flat
}

// ---------------- yg = bf16((y + x_in * D) * silu(z)) ----------------
__global__ __launch_bounds__(256)
void gate_kernel(const float* __restrict__ y, const float* __restrict__ xz,
                 const float* __restrict__ Dv, __bf16* __restrict__ yg) {
  int idx = blockIdx.x * 256 + threadIdx.x;
  if (idx >= NTOK * D_INNER) return;
  int c   = idx & (D_INNER - 1);
  int row = idx >> 11;
  const float* r = xz + (size_t)row * (2 * D_INNER);
  float xin = r[c];
  float z   = r[D_INNER + c];
  float sz  = z / (1.f + __expf(-z));
  yg[idx] = (__bf16)((y[idx] + xin * Dv[c]) * sz);
}

// ---------------------------------------------------------------------------
extern "C" void kernel_launch(void* const* d_in, const int* in_sizes, int n_in,
                              void* d_out, int out_size, void* d_ws, size_t ws_size,
                              hipStream_t stream) {
  (void)in_sizes; (void)n_in; (void)out_size; (void)ws_size;
  const float* x      = (const float*)d_in[0];
  const float* W_in   = (const float*)d_in[1];
  const float* conv_w = (const float*)d_in[2];
  const float* conv_b = (const float*)d_in[3];
  const float* W_x    = (const float*)d_in[4];
  const float* W_dt   = (const float*)d_in[5];
  const float* b_dt   = (const float*)d_in[6];
  const float* A_log  = (const float*)d_in[7];
  const float* Dvec   = (const float*)d_in[8];
  const float* W_out  = (const float*)d_in[9];

  float* out    = (float*)d_out;                       // (B, S, D_MODEL)
  float* h_last = out + (size_t)NTOK * D_MODEL;        // (B, D_INNER, D_STATE)

  // ---- workspace layout (256B-aligned regions) ----
  char* p = (char*)d_ws;
  auto take = [&](size_t bytes) -> char* {
    char* r = p; p += (bytes + 255) & ~(size_t)255; return r;
  };
  __bf16* x_bf   = (__bf16*)take((size_t)NTOK * D_MODEL * 2);         // 4 MB
  __bf16* Wint   = (__bf16*)take((size_t)(2*D_INNER) * D_MODEL * 2);  // 8 MB, (4096 x 1024)
  __bf16* Wxt    = (__bf16*)take((size_t)P_LD * D_INNER * 2);         // (128 x 2048), padded
  __bf16* Wdtt   = (__bf16*)take((size_t)D_INNER * DT_RANK * 2);      // (2048 x 64)
  __bf16* Woutt  = (__bf16*)take((size_t)D_MODEL * D_INNER * 2);      // 4 MB, (1024 x 2048)
  float*  xz     = (float*) take((size_t)NTOK * 2 * D_INNER * 4);     // 32 MB
  float*  xa     = (float*) take((size_t)NTOK * D_INNER * 4);         // 16 MB
  float*  ssm    = (float*) take((size_t)NTOK * P_LD * 4);            // padded, 1 MB
  float*  dt     = (float*) take((size_t)NTOK * D_INNER * 4);         // 16 MB
  float*  yv     = (float*) take((size_t)NTOK * D_INNER * 4);         // 16 MB
  __bf16* yg     = (__bf16*)take((size_t)NTOK * D_INNER * 2);         // 8 MB
  // safe aliases (producer finished before reuse):
  __bf16* xa_bf  = Wint;   // Wint dead after GEMM1; same size
  __bf16* ssm_bf = x_bf;   // x_bf dead after GEMM1; 512 KB <= 4 MB

  // ---- 1) convert x; transpose+convert all GEMM B-operands (weights) ----
  cvt_f32_bf16x4_kernel<<<768, 256, 0, stream>>>(x_bf, x, NTOK * D_MODEL / 4);
  transpose_cvt_kernel<<<dim3((2*D_INNER)/32, D_MODEL/32), 256, 0, stream>>>(
      Wint, W_in, D_MODEL, 2 * D_INNER);                 // (1024 x 4096) -> (4096 x 1024)
  transpose_cvt_kernel<<<dim3(P_LD/32, D_INNER/32), 256, 0, stream>>>(
      Wxt, W_x, D_INNER, P_DIM);                         // (2048 x 96) -> (128 x 2048), pad
  transpose_cvt_kernel<<<dim3(D_INNER/32, DT_RANK/32), 256, 0, stream>>>(
      Wdtt, W_dt, DT_RANK, D_INNER);                     // (64 x 2048) -> (2048 x 64)
  transpose_cvt_kernel<<<dim3(D_MODEL/32, D_INNER/32), 256, 0, stream>>>(
      Woutt, W_out, D_INNER, D_MODEL);                   // (2048 x 1024) -> (1024 x 2048)

  // ---- 2) xz = x @ W_in : (2048 x 1024) x (1024 x 4096) ----
  {
    dim3 g((2 * D_INNER) / BN, NTOK / BM);
    gemm_bf16_wmma_kernel<<<g, 256, 0, stream>>>(x_bf, D_MODEL, Wint, D_MODEL,
                                                 xz, 2 * D_INNER,
                                                 NTOK, 2 * D_INNER, D_MODEL);
  }

  // ---- 3) causal depthwise conv + SiLU ----
  conv_silu_kernel<<<(NTOK * D_INNER) / 256, 256, 0, stream>>>(xz, conv_w, conv_b, xa, xa_bf);

  // ---- 4) ssm_p = x_act @ W_x : (2048 x 2048) x (2048 x 128 padded) ----
  {
    dim3 g(P_LD / BN, NTOK / BM);
    gemm_bf16_wmma_kernel<<<g, 256, 0, stream>>>(xa_bf, D_INNER, Wxt, D_INNER,
                                                 ssm, P_LD,
                                                 NTOK, P_LD, D_INNER);
  }
  cvt_f32_bf16x4_kernel<<<768, 256, 0, stream>>>(ssm_bf, ssm, NTOK * P_LD / 4);

  // ---- 5) dt_lin = dt_r @ W_dt : (2048 x 64, lda 128) x (64 x 2048) ----
  {
    dim3 g(D_INNER / BN, NTOK / BM);
    gemm_bf16_wmma_kernel<<<g, 256, 0, stream>>>(ssm_bf, P_LD, Wdtt, DT_RANK,
                                                 dt, D_INNER,
                                                 NTOK, D_INNER, DT_RANK);
  }
  softplus_bias_kernel<<<(NTOK * D_INNER) / 256, 256, 0, stream>>>(dt, b_dt);

  // ---- 6) selective scan -> y, h_last ----
  scan_kernel<<<(BATCH * D_INNER * D_STATE) / 256, 256, 0, stream>>>(dt, xa, ssm, A_log,
                                                                     yv, h_last);

  // ---- 7) gate: yg = bf16((y + x_in*D) * silu(z)) ----
  gate_kernel<<<(NTOK * D_INNER) / 256, 256, 0, stream>>>(yv, xz, Dvec, yg);

  // ---- 8) out = yg @ W_out : (2048 x 2048) x (2048 x 1024) ----
  {
    dim3 g(D_MODEL / BN, NTOK / BM);
    gemm_bf16_wmma_kernel<<<g, 256, 0, stream>>>(yg, D_INNER, Woutt, D_INNER,
                                                 out, D_MODEL,
                                                 NTOK, D_MODEL, D_INNER);
  }
}
